// Model_SIGNNET_57148834840918
// MI455X (gfx1250) — matrix operand order repaired and checked by reading, not compiled, not run on hardware
//
#include <hip/hip_runtime.h>
#include <hip/hip_bf16.h>

// ---------------- problem constants (from reference) ----------------
#define N_E_    100000
#define N_I_    100000
#define N_TOT_  200000
#define K_      4
#define E_      800000
#define E_HOM_  1600000
#define C_      128
#define F_      64
#define B_      512

#define CDIV(a,b) (((a)+(b)-1)/(b))

typedef __attribute__((ext_vector_type(16))) _Float16 v16h;
typedef __attribute__((ext_vector_type(8)))  _Float16 v8h;
typedef __attribute__((ext_vector_type(8)))  float    v8f;

// =====================================================================
//  WMMA GEMM  out[M x 128] = A1[M x K1] @ B1 (+ A2[M x K2] @ B2) + bias
//             (+ addv) with optional ReLU.  f16 operands, f32 accumulate.
//  Block = 256 threads = 8 waves; one wave -> one 16x16 tile of the
//  N=128 strip.  The 16 x K A-tile is staged ONCE per block into LDS
//  (coalesced fp32 loads, converted to f16), then each wave reads its
//  WMMA A-fragments with two ds_load_b128 per k-step.  B matrices are
//  pre-packed into WMMA B-fragment order (pack_weight_f16) so the B
//  feed is one contiguous 32B load per lane per k-step.
// =====================================================================

// LDS row stride = K + 8 halves (16B pad) -> lane lm hits banks 4*lm mod 64.
template<int K>
__device__ __forceinline__ void stage_A(const float* __restrict__ A,
                                        _Float16* sA, int m0, int tid)
{
  constexpr int CH  = 16 * K / 4;     // float4 chunks in the tile
  constexpr int PER = CH / 256;       // chunks per thread (K=64:1, K=128:2)
#pragma unroll
  for (int i = 0; i < PER; ++i) {
    const int c   = tid + i * 256;
    const int row = c / (K / 4);
    const int kp  = (c % (K / 4)) * 4;
    const float4 f = *(const float4*)(A + (size_t)(m0 + row) * K + kp);
    _Float16* d = sA + row * (K + 8) + kp;
    d[0] = (_Float16)f.x; d[1] = (_Float16)f.y;
    d[2] = (_Float16)f.z; d[3] = (_Float16)f.w;
  }
}

template<int K>
__device__ __forceinline__ v8f accum_tile(v8f acc, const _Float16* sA,
                                          const _Float16* __restrict__ Bp,
                                          int lane, int nt)
{
  const int lg = lane >> 4, lm = lane & 15;
#pragma unroll
  for (int k0 = 0; k0 < K; k0 += 32) {
    // dense 16-bit A 16x32 fragment: element e -> K = k0 + e + 16*lg
    const _Float16* pa = sA + lm * (K + 8) + k0 + 16 * lg;
    const v8h a0 = *(const v8h*)pa;          // ds_load_b128
    const v8h a1 = *(const v8h*)(pa + 8);    // ds_load_b128
    v16h a;
#pragma unroll
    for (int i = 0; i < 8; ++i) { a[i] = a0[i]; a[8 + i] = a1[i]; }
    // dense 16-bit B 32x16 fragment, pre-swizzled: one 32B load.
    const v16h b = *(const v16h*)(Bp + ((size_t)(((k0 >> 5) * 8 + nt) * 32 + lane) << 4));
    acc = __builtin_amdgcn_wmma_f32_16x16x32_f16(false, a, false, b,
                                                 (short)0, acc, false, false);
  }
  return acc;
}

template<int K1, int K2, bool HAS_ADD, bool RELU>
__global__ void __launch_bounds__(256)
gemm128(const float* __restrict__ A1, const _Float16* __restrict__ B1,
        const float* __restrict__ A2, const _Float16* __restrict__ B2,
        const float* __restrict__ bias, const float* __restrict__ addv,
        float* __restrict__ out)
{
  __shared__ _Float16 sA[16 * (128 + 8)];   // 4352 B, covers K <= 128
  const int tid  = threadIdx.x;
  const int nt   = tid >> 5;                // n-tile 0..7 (N = 128)
  const int lane = tid & 31;
  const int lg = lane >> 4, lm = lane & 15;
  const int m0 = blockIdx.x << 4;

  stage_A<K1>(A1, sA, m0, tid);
  __syncthreads();
  v8f acc = {};
  acc = accum_tile<K1>(acc, sA, B1, lane, nt);
  if constexpr (K2 > 0) {
    __syncthreads();
    stage_A<K2>(A2, sA, m0, tid);
    __syncthreads();
    acc = accum_tile<K2>(acc, sA, B2, lane, nt);
  }

  const int col = (nt << 4) + lm;
  const float bv = bias[col];
#pragma unroll
  for (int r = 0; r < 8; ++r) {
    // D layout: VGPR r, lane l -> M = r + 8*(l/16), N = l%16
    const int orow = m0 + r + (lg << 3);
    float v = acc[r] + bv;
    if constexpr (HAS_ADD) v += addv[(size_t)orow * C_ + col];
    if constexpr (RELU)    v = fmaxf(v, 0.f);
    out[(size_t)orow * C_ + col] = v;
  }
}

// Pack fp32 weight W[K x 128] into f16 WMMA B-fragment order:
// P[((kt*8+nt)*32 + lane)*16 + e] = W[kt*32 + e + 16*(lane/16)][nt*16 + lane%16]
__global__ void pack_weight_f16(const float* __restrict__ W,
                                _Float16* __restrict__ P, int K)
{
  int idx = blockIdx.x * blockDim.x + threadIdx.x;
  if (idx >= K * C_) return;
  const int e    = idx & 15;
  const int lane = (idx >> 4) & 31;
  const int t    = idx >> 9;          // 512 elems per 32x16 tile
  const int nt   = t & 7;
  const int kt   = t >> 3;
  const int lg = lane >> 4, ln = lane & 15;
  const int k = kt * 32 + e + 16 * lg;   // dense 16-bit B 32x16 layout
  const int n = nt * 16 + ln;
  P[idx] = (_Float16)W[(size_t)k * C_ + n];
}

// =====================================================================
//  GIN / PE kernels
// =====================================================================

__global__ void init_h(float* __restrict__ h, const float* __restrict__ V,
                       float sg, int n)
{
  int i = blockIdx.x * blockDim.x + threadIdx.x;
  if (i < n) h[(size_t)i * 16] = sg * V[i];
}

// one thread per (edge, k, channel): single gather + single atomic each
template<int D>
__global__ void gin_agg(const float* __restrict__ h, float* __restrict__ agg,
                        const int* __restrict__ src, const int* __restrict__ dst,
                        int total)   // total = E_HOM * 4 * D
{
  int idx = blockIdx.x * blockDim.x + threadIdx.x;
  if (idx >= total) return;
  const int c  = idx % D;
  const int jk = idx / D;
  const int j = jk >> 2, k = jk & 3;
  atomicAdd(agg + ((size_t)dst[j] * 4 + k) * 16 + c,
            h[((size_t)src[j] * 4 + k) * 16 + c]);
}

template<int DIN, int DOUT, bool RELU>
__global__ void gin_mlp(float* h, const float* __restrict__ agg,
                        const float* __restrict__ W0, const float* __restrict__ b0,
                        const float* __restrict__ W1, const float* __restrict__ b1,
                        int nrows)
{
  int row = blockIdx.x * blockDim.x + threadIdx.x;
  if (row >= nrows) return;
  float* hr = h + (size_t)row * 16;
  const float* ar = agg + (size_t)row * 16;
  float z[DIN];
#pragma unroll
  for (int c = 0; c < DIN; ++c) z[c] = hr[c] + ar[c];
  float y[16];
#pragma unroll
  for (int j = 0; j < 16; ++j) {
    float s = b0[j];
#pragma unroll
    for (int c = 0; c < DIN; ++c) s += z[c] * W0[c * 16 + j];
    y[j] = fmaxf(s, 0.f);
  }
#pragma unroll
  for (int t = 0; t < DOUT; ++t) {
    float s = b1[t];
#pragma unroll
    for (int j = 0; j < 16; ++j) s += y[j] * W1[j * DOUT + t];
    if (RELU) s = fmaxf(s, 0.f);
    hr[t] = s;
  }
}

__global__ void pe4_accum(const float* __restrict__ h, float* __restrict__ pe4, int n)
{
  int idx = blockIdx.x * blockDim.x + threadIdx.x;
  if (idx >= n) return;
  const int node = idx >> 2, c = idx & 3;
  const float* hp = h + (size_t)node * 64;   // 4 k-rows x 16 floats
  pe4[idx] += hp[c] + hp[16 + c] + hp[32 + c] + hp[48 + c];
}

__global__ void rho_mlp(const float* __restrict__ pe4, float* __restrict__ pe16,
                        const float* __restrict__ W0, const float* __restrict__ b0,
                        const float* __restrict__ W1, const float* __restrict__ b1,
                        int n)
{
  int i = blockIdx.x * blockDim.x + threadIdx.x;
  if (i >= n) return;
  const float t0 = pe4[i * 4 + 0], t1 = pe4[i * 4 + 1];
  const float t2 = pe4[i * 4 + 2], t3 = pe4[i * 4 + 3];
  float v[16];
#pragma unroll
  for (int j = 0; j < 16; ++j) v[j] = b1[j];
  for (int hh = 0; hh < 64; ++hh) {
    float u = b0[hh] + t0 * W0[hh] + t1 * W0[64 + hh]
                     + t2 * W0[128 + hh] + t3 * W0[192 + hh];
    u = fmaxf(u, 0.f);
#pragma unroll
    for (int j = 0; j < 16; ++j) v[j] += u * W1[hh * 16 + j];
  }
#pragma unroll
  for (int j = 0; j < 16; ++j) pe16[(size_t)i * 16 + j] = v[j];
}

__global__ void pe_proj(const float* __restrict__ pe16, float* __restrict__ pe,
                        const float* __restrict__ W, const float* __restrict__ b,
                        int total)
{
  int idx = blockIdx.x * blockDim.x + threadIdx.x;
  if (idx >= total) return;
  const int node = idx >> 7, c = idx & 127;
  const float* p = pe16 + (size_t)node * 16;
  float s = b[c];
#pragma unroll
  for (int j = 0; j < 16; ++j) s += p[j] * W[j * C_ + c];
  pe[idx] = s;
}

// =====================================================================
//  Time features, SAGE aggregation, head
// =====================================================================

__global__ void time_feat(const float* __restrict__ seed_t,
                          const int* __restrict__ batch,
                          const float* __restrict__ t_ent,
                          float* __restrict__ tf, int total)
{
  int idx = blockIdx.x * blockDim.x + threadIdx.x;
  if (idx >= total) return;
  const int i = idx >> 7, j = idx & 127;
  const float rel  = seed_t[batch[i]] - t_ent[i];
  const float freq = expf(-(float)j * (9.210340371976184f / 128.f)); // ln(1e4)/C
  tf[idx] = cosf(rel * freq);
}

// one thread per (edge, channel): single gather + single atomic
__global__ void sage_scatter(const float* __restrict__ x, float* __restrict__ agg,
                             const int* __restrict__ src, const int* __restrict__ dst,
                             int total)   // total = E * 128
{
  int idx = blockIdx.x * blockDim.x + threadIdx.x;
  if (idx >= total) return;
  const int j = idx >> 7, c = idx & 127;
  atomicAdd(agg + (size_t)dst[j] * C_ + c, x[(size_t)src[j] * C_ + c]);
}

__global__ void edge_count(float* __restrict__ cnt, const int* __restrict__ dst, int e)
{
  int j = blockIdx.x * blockDim.x + threadIdx.x;
  if (j < e) atomicAdd(cnt + dst[j], 1.f);
}

__global__ void mean_norm(float* __restrict__ agg, const float* __restrict__ cnt, int total)
{
  int idx = blockIdx.x * blockDim.x + threadIdx.x;
  if (idx < total) agg[idx] /= fmaxf(cnt[idx >> 7], 1.f);
}

__global__ void head_k(const float* __restrict__ xe, const float* __restrict__ W,
                       const float* __restrict__ b, float* __restrict__ out, int nb)
{
  int i = blockIdx.x * blockDim.x + threadIdx.x;
  if (i >= nb) return;
  float s = b[0];
  const float* xp = xe + (size_t)i * C_;
#pragma unroll 8
  for (int c = 0; c < C_; ++c) s += xp[c] * W[c];
  out[i] = s;
}

// =====================================================================
//  Launcher
// =====================================================================

extern "C" void kernel_launch(void* const* d_in, const int* in_sizes, int n_in,
                              void* d_out, int out_size, void* d_ws, size_t ws_size,
                              hipStream_t stream)
{
  (void)in_sizes; (void)n_in; (void)out_size; (void)ws_size;

  const float* V       = (const float*)d_in[0];
  const float* x_ent   = (const float*)d_in[1];
  const float* x_itm   = (const float*)d_in[2];
  const float* t_ent   = (const float*)d_in[3];
  const float* seed_t  = (const float*)d_in[4];
  const int*   hom_src = (const int*)d_in[5];
  const int*   hom_dst = hom_src + E_HOM_;
  const int*   ei_src  = (const int*)d_in[6];
  const int*   ei_dst  = ei_src + E_;
  const int*   ie_src  = (const int*)d_in[7];
  const int*   ie_dst  = ie_src + E_;
  const int*   batch_e = (const int*)d_in[8];

  // params in _init_params insertion order
  const float* prm[38];
  for (int i = 0; i < 38; ++i) prm[i] = (const float*)d_in[9 + i];
  // 0..3 gin0{W0,b0,W1,b1}  4..7 gin1  8..11 gin2
  // 12..15 rho{W0,b0,W1,b1} 16 pe_W 17 pe_b
  // 18/19 enc_e 20/21 enc_i 22/23 time
  // 24..29 sage0: self_e,nbr_e,b_e,self_i,nbr_i,b_i   30..35 sage1 same
  // 36 head_W 37 head_b

  // ---------------- workspace layout (aliased across phases) ----------------
  char* w = (char*)d_ws;
  auto alloc = [&](size_t bytes) -> char* {
    char* p = w; w += (bytes + 255) & ~(size_t)255; return p;
  };
  float* bufA = (float*)alloc((size_t)N_TOT_ * K_ * 16 * 4); // GIN h   -> x_e ping
  float* bufB = (float*)alloc((size_t)N_TOT_ * K_ * 16 * 4); // GIN agg -> x_i ping
  float* peB  = (float*)alloc((size_t)N_TOT_ * C_ * 4);      // pe      -> x_e/x_i pong
  float* scrC = (float*)alloc((size_t)N_E_ * C_ * 4);        // pe4/pe16-> agg_e
  float* scrD = (float*)alloc((size_t)N_E_ * C_ * 4);        // tf      -> agg_i
  float* cntE = (float*)alloc((size_t)N_E_ * 4);
  float* cntI = (float*)alloc((size_t)N_I_ * 4);
  _Float16* pk_enc_e = (_Float16*)alloc((size_t)F_ * C_ * 2);
  _Float16* pk_enc_i = (_Float16*)alloc((size_t)F_ * C_ * 2);
  _Float16* pk_time  = (_Float16*)alloc((size_t)C_ * C_ * 2);
  _Float16* pk_sage[8];
  for (int i = 0; i < 8; ++i) pk_sage[i] = (_Float16*)alloc((size_t)C_ * C_ * 2);

  // ---------------- pack all WMMA B operands to f16 fragments ----------------
  pack_weight_f16<<<CDIV(F_ * C_, 256), 256, 0, stream>>>(prm[18], pk_enc_e, F_);
  pack_weight_f16<<<CDIV(F_ * C_, 256), 256, 0, stream>>>(prm[20], pk_enc_i, F_);
  pack_weight_f16<<<CDIV(C_ * C_, 256), 256, 0, stream>>>(prm[22], pk_time,  C_);
  const int sage_w_idx[8] = {24, 25, 27, 28, 30, 31, 33, 34}; // se,ne,si,ni per layer
  for (int i = 0; i < 8; ++i)
    pack_weight_f16<<<CDIV(C_ * C_, 256), 256, 0, stream>>>(prm[sage_w_idx[i]],
                                                            pk_sage[i], C_);

  // ---------------- SignNet positional encoding ----------------
  float* pe4  = scrC;
  float* pe16 = scrC + (size_t)N_TOT_ * 4;
  hipMemsetAsync(pe4, 0, (size_t)N_TOT_ * 4 * sizeof(float), stream);

  const int NK  = N_TOT_ * K_;       // 800k gin rows
  const int NEK = E_HOM_ * K_;       // 6.4M (edge,k) pairs
  for (int sign = 0; sign < 2; ++sign) {
    init_h<<<CDIV(NK, 256), 256, 0, stream>>>(bufA, V, sign ? -1.f : 1.f, NK);
    // layer 0: 1 -> 16 -> 16, relu
    hipMemsetAsync(bufB, 0, (size_t)NK * 16 * sizeof(float), stream);
    gin_agg<1><<<CDIV(NEK, 256), 256, 0, stream>>>(bufA, bufB, hom_src, hom_dst, NEK);
    gin_mlp<1, 16, true><<<CDIV(NK, 256), 256, 0, stream>>>(
        bufA, bufB, prm[0], prm[1], prm[2], prm[3], NK);
    // layer 1: 16 -> 16 -> 16, relu
    hipMemsetAsync(bufB, 0, (size_t)NK * 16 * sizeof(float), stream);
    gin_agg<16><<<CDIV(NEK * 16, 256), 256, 0, stream>>>(bufA, bufB, hom_src, hom_dst,
                                                         NEK * 16);
    gin_mlp<16, 16, true><<<CDIV(NK, 256), 256, 0, stream>>>(
        bufA, bufB, prm[4], prm[5], prm[6], prm[7], NK);
    // layer 2: 16 -> 16 -> 4, no final relu
    hipMemsetAsync(bufB, 0, (size_t)NK * 16 * sizeof(float), stream);
    gin_agg<16><<<CDIV(NEK * 16, 256), 256, 0, stream>>>(bufA, bufB, hom_src, hom_dst,
                                                         NEK * 16);
    gin_mlp<16, 4, false><<<CDIV(NK, 256), 256, 0, stream>>>(
        bufA, bufB, prm[8], prm[9], prm[10], prm[11], NK);
    pe4_accum<<<CDIV(N_TOT_ * 4, 256), 256, 0, stream>>>(bufA, pe4, N_TOT_ * 4);
  }
  rho_mlp<<<CDIV(N_TOT_, 256), 256, 0, stream>>>(pe4, pe16, prm[12], prm[13],
                                                 prm[14], prm[15], N_TOT_);
  pe_proj<<<CDIV(N_TOT_ * C_, 256), 256, 0, stream>>>(pe16, peB, prm[16], prm[17],
                                                      N_TOT_ * C_);

  // ---------------- encoders + time embedding (WMMA) ----------------
  gemm128<F_, 0, true, false><<<N_E_ / 16, 256, 0, stream>>>(
      x_ent, pk_enc_e, nullptr, nullptr, prm[19], peB, bufA);
  gemm128<F_, 0, true, false><<<N_I_ / 16, 256, 0, stream>>>(
      x_itm, pk_enc_i, nullptr, nullptr, prm[21], peB + (size_t)N_E_ * C_, bufB);
  time_feat<<<CDIV(N_E_ * C_, 256), 256, 0, stream>>>(seed_t, batch_e, t_ent,
                                                      scrD, N_E_ * C_);
  gemm128<C_, 0, true, false><<<N_E_ / 16, 256, 0, stream>>>(
      scrD, pk_time, nullptr, nullptr, prm[23], bufA, bufA);

  // ---------------- 2x heteroSAGE layers ----------------
  float* cur_e = bufA; float* cur_i = bufB;
  float* nxt_e = peB;  float* nxt_i = peB + (size_t)N_E_ * C_;
  const int sage_b_e[2] = {26, 32};
  const int sage_b_i[2] = {29, 35};
  for (int l = 0; l < 2; ++l) {
    // agg_i = mean over edge_ei of cur_e  -> scrD
    hipMemsetAsync(scrD, 0, (size_t)N_I_ * C_ * sizeof(float), stream);
    hipMemsetAsync(cntI, 0, (size_t)N_I_ * sizeof(float), stream);
    sage_scatter<<<CDIV(E_ * 128, 256), 256, 0, stream>>>(cur_e, scrD,
                                                          ei_src, ei_dst, E_ * 128);
    edge_count<<<CDIV(E_, 256), 256, 0, stream>>>(cntI, ei_dst, E_);
    mean_norm<<<CDIV(N_I_ * C_, 256), 256, 0, stream>>>(scrD, cntI, N_I_ * C_);
    // agg_e = mean over edge_ie of cur_i  -> scrC
    hipMemsetAsync(scrC, 0, (size_t)N_E_ * C_ * sizeof(float), stream);
    hipMemsetAsync(cntE, 0, (size_t)N_E_ * sizeof(float), stream);
    sage_scatter<<<CDIV(E_ * 128, 256), 256, 0, stream>>>(cur_i, scrC,
                                                          ie_src, ie_dst, E_ * 128);
    edge_count<<<CDIV(E_, 256), 256, 0, stream>>>(cntE, ie_dst, E_);
    mean_norm<<<CDIV(N_E_ * C_, 256), 256, 0, stream>>>(scrC, cntE, N_E_ * C_);
    // fused dual-GEMM updates: relu(x@self + agg@nbr + b)
    gemm128<C_, C_, false, true><<<N_E_ / 16, 256, 0, stream>>>(
        cur_e, pk_sage[l * 4 + 0], scrC, pk_sage[l * 4 + 1],
        prm[sage_b_e[l]], nullptr, nxt_e);
    gemm128<C_, C_, false, true><<<N_I_ / 16, 256, 0, stream>>>(
        cur_i, pk_sage[l * 4 + 2], scrD, pk_sage[l * 4 + 3],
        prm[sage_b_i[l]], nullptr, nxt_i);
    float* t0 = cur_e; cur_e = nxt_e; nxt_e = t0;
    float* t1 = cur_i; cur_i = nxt_i; nxt_i = t1;
  }

  // ---------------- head ----------------
  head_k<<<CDIV(B_, 256), 256, 0, stream>>>(cur_e, prm[36], prm[37],
                                            (float*)d_out, B_);
}
